// AgentAttention_9938554323541
// MI455X (gfx1250) — compile-verified, matrix operand order checked by reading, not run on hardware
//
#include <hip/hip_runtime.h>
#include <hip/hip_bf16.h>

typedef __attribute__((ext_vector_type(16))) _Float16 v16h;
typedef __attribute__((ext_vector_type(8)))  float    v8f;

// ---------------------------------------------------------------------------
// WMMA helpers (gfx1250, wave32). A: 16x32 f16 (row-striped per lane),
// B: 32x16 f16 (column n contiguous over K in memory), C/D: 16x16 f32.
// Per ISA 7.12.2:  A lane l (0..31): row = l&15, K = (l>>4)*8 + e (e<8)
//                                          and 16 + (l>>4)*8 + (e-8) (e>=8)
// B loaded with the symmetric layout (col = l&15, same K striping).
// C: lane l -> col = l&15, row = (l>>4)*8 + r  (r = accumulator index 0..7)
// ---------------------------------------------------------------------------
__device__ __forceinline__ v16h load_frag(const _Float16* __restrict__ p, int ld) {
  const int l  = threadIdx.x & 31;
  const int r  = l & 15;
  const int hi = l >> 4;
  const _Float16* b = p + r * ld + hi * 8;
  v16h f;
#pragma unroll
  for (int e = 0; e < 8; ++e) { f[e] = b[e]; f[8 + e] = b[16 + e]; }
  return f;
}

__device__ __forceinline__ v8f wmma16(v16h a, v16h b, v8f c) {
  return __builtin_amdgcn_wmma_f32_16x16x32_f16(false, a, false, b, (short)0, c,
                                                false, false);
}

// ---------------------------------------------------------------------------
// Prep: cast weights + x body to f16
// ---------------------------------------------------------------------------
__global__ void k_prep_w(const float* __restrict__ qkvw, const float* __restrict__ projw,
                         _Float16* __restrict__ Wq, _Float16* __restrict__ Wp) {
  int idx = blockIdx.x * 256 + threadIdx.x;            // 768 blocks
  if (idx < 768 * 256) Wq[idx] = (_Float16)qkvw[idx];
  if (idx < 256 * 256) Wp[idx] = (_Float16)projw[idx];
}

__global__ void k_cast_x(const float* __restrict__ x, _Float16* __restrict__ Xh) {
  int idx = blockIdx.x * 256 + threadIdx.x;            // 32768 blocks
  int cc = idx & 255, row = idx >> 8;                  // row = b*1024 + j
  int b = row >> 10, j = row & 1023;
  Xh[idx] = (_Float16)x[(b * 1025 + 1 + j) * 256 + cc];
}

// ---------------------------------------------------------------------------
// qkv GEMM: [32768,256] x [256,768]; wave owns 16x64 tile, K-loop 8 steps.
// All fragment loads of a k-step are issued before the 4 WMMAs so the
// s_wait_loadcnt values stagger (overlap loads with matrix math).
// ---------------------------------------------------------------------------
__global__ void k_gemm_qkv(const _Float16* __restrict__ Xh, const _Float16* __restrict__ W,
                           float* __restrict__ Qf, float* __restrict__ Vf,
                           _Float16* __restrict__ Qh, _Float16* __restrict__ Kh,
                           _Float16* __restrict__ Vt) {
  const int wid = blockIdx.x * 8 + (threadIdx.x >> 5);   // 24576 waves
  const int mt = wid / 12, nc = wid % 12;
  const int m0 = mt * 16, n0 = nc * 64;
  v8f acc[4] = {};
#pragma unroll
  for (int k0 = 0; k0 < 256; k0 += 32) {
    v16h a = load_frag(Xh + m0 * 256 + k0, 256);
    v16h b[4];
#pragma unroll
    for (int t = 0; t < 4; ++t)
      b[t] = load_frag(W + (n0 + t * 16) * 256 + k0, 256);
#pragma unroll
    for (int t = 0; t < 4; ++t)
      acc[t] = wmma16(a, b[t], acc[t]);
  }
  const int l = threadIdx.x & 31, nn = l & 15, mb = (l >> 4) << 3;
#pragma unroll
  for (int t = 0; t < 4; ++t) {
    int ng = n0 + t * 16 + nn;
    int sec = ng >> 8, d = ng & 255, h = d >> 5, dd = d & 31;
#pragma unroll
    for (int r = 0; r < 8; ++r) {
      int mg = m0 + mb + r;
      int bi = mg >> 10, j = mg & 1023;
      float v = acc[t][r];
      if (sec == 0) {                       // Q: f32 for pooling + f16 [b,h,n,hd]
        Qf[mg * 256 + d] = v;
        Qh[((bi * 8 + h) * 1024 + j) * 32 + dd] = (_Float16)v;
      } else if (sec == 1) {                // K: f16 [b,h,n,hd]
        Kh[((bi * 8 + h) * 1024 + j) * 32 + dd] = (_Float16)v;
      } else {                              // V: f32 for dwc + f16 [b,h,hd,n]
        Vf[mg * 256 + d] = v;
        Vt[((bi * 8 + h) * 32 + dd) * 1024 + j] = (_Float16)v;
      }
    }
  }
}

// ---------------------------------------------------------------------------
// Adaptive avg pool 32x32 -> 7x7 (exact rectangle means), scale folded in.
// Ah: f16 [b, h, 64(pad), 32]; rows >= 49 zeroed.
// ---------------------------------------------------------------------------
__global__ void k_agents(const float* __restrict__ Qf, _Float16* __restrict__ Ah) {
  int idx = blockIdx.x * 256 + threadIdx.x;   // 32*64*256 = 524288
  int ch = idx & 255;
  int a  = (idx >> 8) & 63;
  int b  = idx >> 14;
  int h = ch >> 5, dd = ch & 31;
  float out = 0.f;
  if (a < 49) {
    int pi = a / 7, pj = a % 7;
    int s0 = pi * 32 / 7, e0 = ((pi + 1) * 32 + 6) / 7;
    int s1 = pj * 32 / 7, e1 = ((pj + 1) * 32 + 6) / 7;
    float s = 0.f;
    for (int i = s0; i < e0; ++i)
      for (int j = s1; j < e1; ++j)
        s += Qf[((b << 10) + i * 32 + j) * 256 + ch];
    out = s / (float)((e0 - s0) * (e1 - s1)) * 0.17677669529663687f; // hd^-0.5
  }
  Ah[((b * 8 + h) * 64 + a) * 32 + dd] = (_Float16)out;
}

// ---------------------------------------------------------------------------
// Bilinear 7x7 -> 32x32 (half-pixel centers, edge clamp)
// ---------------------------------------------------------------------------
__device__ __forceinline__ float bilin7to32(const float* __restrict__ img, int i, int j) {
  float fy = (i + 0.5f) * 0.21875f - 0.5f;
  float fx = (j + 0.5f) * 0.21875f - 0.5f;
  float yf = floorf(fy), xf = floorf(fx);
  float wy = fy - yf, wx = fx - xf;
  int y0 = (int)yf, x0 = (int)xf;
  int y1 = min(6, max(0, y0 + 1)), x1 = min(6, max(0, x0 + 1));
  y0 = min(6, max(0, y0)); x0 = min(6, max(0, x0));
  float v00 = img[y0 * 7 + x0], v01 = img[y0 * 7 + x1];
  float v10 = img[y1 * 7 + x0], v11 = img[y1 * 7 + x1];
  return (1.f - wy) * ((1.f - wx) * v00 + wx * v01) +
         wy * ((1.f - wx) * v10 + wx * v11);
}

__global__ void k_biasA(const float* __restrict__ anb, const float* __restrict__ ahb,
                        const float* __restrict__ awb, float* __restrict__ BA) {
  int idx = blockIdx.x * 256 + threadIdx.x;   // [h,64,1024]
  int n = idx & 1023, a = (idx >> 10) & 63, h = idx >> 16;
  float v = 0.f;
  if (a < 49) {
    int i = n >> 5, j = n & 31;
    v = bilin7to32(anb + (h * 49 + a) * 49, i, j)
      + ahb[(h * 49 + a) * 32 + i] + awb[(h * 49 + a) * 32 + j];
  }
  BA[idx] = v;
}

__global__ void k_biasQ(const float* __restrict__ nab, const float* __restrict__ hab,
                        const float* __restrict__ wab, float* __restrict__ BQ) {
  int idx = blockIdx.x * 256 + threadIdx.x;   // [h,1024,64]
  int a = idx & 63, n = (idx >> 6) & 1023, h = idx >> 16;
  float v = -1e30f;                            // padded agents -> prob 0
  if (a < 49) {
    int i = n >> 5, j = n & 31;
    v = bilin7to32(nab + (h * 49 + a) * 49, i, j)
      + hab[(h * 32 + i) * 49 + a] + wab[(h * 32 + j) * 49 + a];
  }
  BQ[idx] = v;
}

// ---------------------------------------------------------------------------
// S1[b,h,64,1024] = Ah @ Kh^T + biasA   (K = 32, one wmma per 16x16 tile)
// ---------------------------------------------------------------------------
__global__ void k_gemm_s1(const _Float16* __restrict__ Ah, const _Float16* __restrict__ Kh,
                          const float* __restrict__ BA, float* __restrict__ S) {
  const int wid = blockIdx.x * 8 + (threadIdx.x >> 5);   // 16384 waves
  const int bh = wid >> 6, rest = wid & 63;
  const int mt = rest >> 4, nc = rest & 15;
  const int m0 = mt * 16, n0 = nc * 64;
  const int h = bh & 7;
  v16h a = load_frag(Ah + (bh * 64 + m0) * 32, 32);
  v16h b[4];
#pragma unroll
  for (int t = 0; t < 4; ++t)
    b[t] = load_frag(Kh + (bh * 1024 + n0 + t * 16) * 32, 32);
  v8f acc[4] = {};
#pragma unroll
  for (int t = 0; t < 4; ++t)
    acc[t] = wmma16(a, b[t], acc[t]);
  const int l = threadIdx.x & 31, nn = l & 15, mb = (l >> 4) << 3;
#pragma unroll
  for (int t = 0; t < 4; ++t) {
    int ng = n0 + t * 16 + nn;
#pragma unroll
    for (int r = 0; r < 8; ++r) {
      int m = m0 + mb + r;
      S[(bh * 64 + m) * 1024 + ng] = acc[t][r] + BA[(h * 64 + m) * 1024 + ng];
    }
  }
}

// softmax over n=1024 per (b,h,agent); padded rows -> zeros
__global__ void k_softmax1(const float* __restrict__ S, _Float16* __restrict__ P) {
  const int bh = blockIdx.x >> 6, a = blockIdx.x & 63;
  const int base = (bh * 64 + a) * 1024;
  const int t = threadIdx.x;
  if (a >= 49) {
    for (int i = t; i < 1024; i += 256) P[base + i] = (_Float16)0.f;
    return;
  }
  float v[4]; float mx = -3.0e38f;
#pragma unroll
  for (int i = 0; i < 4; ++i) { v[i] = S[base + t + 256 * i]; mx = fmaxf(mx, v[i]); }
#pragma unroll
  for (int d = 16; d > 0; d >>= 1) mx = fmaxf(mx, __shfl_xor(mx, d, 32));
  __shared__ float red[8];
  const int w = t >> 5, l = t & 31;
  if (l == 0) red[w] = mx;
  __syncthreads();
  if (t == 0) { float m = red[0]; for (int i = 1; i < 8; ++i) m = fmaxf(m, red[i]); red[0] = m; }
  __syncthreads();
  mx = red[0];
  float sm = 0.f;
#pragma unroll
  for (int i = 0; i < 4; ++i) { v[i] = __expf(v[i] - mx); sm += v[i]; }
#pragma unroll
  for (int d = 16; d > 0; d >>= 1) sm += __shfl_xor(sm, d, 32);
  __syncthreads();
  if (l == 0) red[w] = sm;
  __syncthreads();
  if (t == 0) { float s2 = 0.f; for (int i = 0; i < 8; ++i) s2 += red[i]; red[0] = s2; }
  __syncthreads();
  const float inv = 1.f / red[0];
#pragma unroll
  for (int i = 0; i < 4; ++i) P[base + t + 256 * i] = (_Float16)(v[i] * inv);
}

// ---------------------------------------------------------------------------
// agent_v^T[b,h,32,64] = (P1[64,1024] @ V[1024,32])^T  (K = 1024, 32 steps)
// ---------------------------------------------------------------------------
__global__ void k_gemm_av(const _Float16* __restrict__ P1, const _Float16* __restrict__ Vt,
                          _Float16* __restrict__ AVt) {
  const int wid = blockIdx.x * 8 + (threadIdx.x >> 5);   // 1024 waves
  const int bh = wid >> 2, mt = wid & 3;
  const int m0 = mt * 16;
  v8f acc[2] = {};
#pragma unroll 4
  for (int k0 = 0; k0 < 1024; k0 += 32) {
    v16h a = load_frag(P1 + (bh * 64 + m0) * 1024 + k0, 1024);
    v16h b[2];
#pragma unroll
    for (int t = 0; t < 2; ++t)
      b[t] = load_frag(Vt + (bh * 32 + t * 16) * 1024 + k0, 1024);
#pragma unroll
    for (int t = 0; t < 2; ++t)
      acc[t] = wmma16(a, b[t], acc[t]);
  }
  const int l = threadIdx.x & 31, nn = l & 15, mb = (l >> 4) << 3;
#pragma unroll
  for (int t = 0; t < 2; ++t) {
    int dd = t * 16 + nn;
#pragma unroll
    for (int r = 0; r < 8; ++r)
      AVt[(bh * 32 + dd) * 64 + (m0 + mb + r)] = (_Float16)acc[t][r];
  }
}

// ---------------------------------------------------------------------------
// S2[b,h,1024,64] = Qh @ Ah^T + biasQ   (scale already inside Ah)
// ---------------------------------------------------------------------------
__global__ void k_gemm_s2(const _Float16* __restrict__ Qh, const _Float16* __restrict__ Ah,
                          const float* __restrict__ BQ, float* __restrict__ S) {
  const int wid = blockIdx.x * 8 + (threadIdx.x >> 5);   // 16384 waves
  const int bh = wid >> 6, mt = wid & 63;
  const int m0 = mt * 16;
  const int h = bh & 7;
  v16h a = load_frag(Qh + (bh * 1024 + m0) * 32, 32);
  v16h b[4];
#pragma unroll
  for (int t = 0; t < 4; ++t)
    b[t] = load_frag(Ah + (bh * 64 + t * 16) * 32, 32);
  v8f acc[4] = {};
#pragma unroll
  for (int t = 0; t < 4; ++t)
    acc[t] = wmma16(a, b[t], acc[t]);
  const int l = threadIdx.x & 31, nn = l & 15, mb = (l >> 4) << 3;
#pragma unroll
  for (int t = 0; t < 4; ++t) {
    int ag = t * 16 + nn;
#pragma unroll
    for (int r = 0; r < 8; ++r) {
      int m = m0 + mb + r;
      S[(bh * 1024 + m) * 64 + ag] = acc[t][r] + BQ[(h * 1024 + m) * 64 + ag];
    }
  }
}

// softmax over 64 (49 valid; rest exp(-1e30)->0), wave per row
__global__ void k_softmax2(const float* __restrict__ S, _Float16* __restrict__ P) {
  const int row = blockIdx.x * 8 + (threadIdx.x >> 5);   // 262144 rows
  const int l = threadIdx.x & 31;
  const float* s = S + (size_t)row * 64;
  float a0 = s[l], a1 = s[l + 32];
  float mx = fmaxf(a0, a1);
#pragma unroll
  for (int d = 16; d > 0; d >>= 1) mx = fmaxf(mx, __shfl_xor(mx, d, 32));
  a0 = __expf(a0 - mx); a1 = __expf(a1 - mx);
  float sm = a0 + a1;
#pragma unroll
  for (int d = 16; d > 0; d >>= 1) sm += __shfl_xor(sm, d, 32);
  const float inv = 1.f / sm;
  _Float16* p = P + (size_t)row * 64;
  p[l] = (_Float16)(a0 * inv);
  p[l + 32] = (_Float16)(a1 * inv);
}

// ---------------------------------------------------------------------------
// O[b,n,c] = P2[1024,64] @ agent_v[64,32]  (per b,h; K = 64, 2 steps)
// ---------------------------------------------------------------------------
__global__ void k_gemm_out(const _Float16* __restrict__ P2, const _Float16* __restrict__ AVt,
                           float* __restrict__ O) {
  const int wid = blockIdx.x * 8 + (threadIdx.x >> 5);   // 16384 waves
  const int bh = wid >> 6, mt = wid & 63;
  const int m0 = mt * 16;
  const int bi = bh >> 3, h = bh & 7;
  v16h a0 = load_frag(P2 + (bh * 1024 + m0) * 64, 64);
  v16h a1 = load_frag(P2 + (bh * 1024 + m0) * 64 + 32, 64);
  v16h b[4];
#pragma unroll
  for (int t = 0; t < 2; ++t) {
    b[t]     = load_frag(AVt + (bh * 32 + t * 16) * 64, 64);
    b[2 + t] = load_frag(AVt + (bh * 32 + t * 16) * 64 + 32, 64);
  }
  v8f acc[2] = {};
#pragma unroll
  for (int t = 0; t < 2; ++t) {
    acc[t] = wmma16(a0, b[t], acc[t]);
    acc[t] = wmma16(a1, b[2 + t], acc[t]);
  }
  const int l = threadIdx.x & 31, nn = l & 15, mb = (l >> 4) << 3;
#pragma unroll
  for (int t = 0; t < 2; ++t) {
    int dd = t * 16 + nn;
#pragma unroll
    for (int r = 0; r < 8; ++r) {
      int m = m0 + mb + r;
      O[((bi << 10) + m) * 256 + h * 32 + dd] = acc[t][r];
    }
  }
}

// depthwise 3x3 SAME on V + bias, added into O
__global__ void k_dwc(const float* __restrict__ Vf, const float* __restrict__ Wd,
                      const float* __restrict__ Bd, float* __restrict__ O) {
  int idx = blockIdx.x * 256 + threadIdx.x;   // 32*1024*256
  int ch = idx & 255;
  int pix = idx >> 8;
  int b = pix >> 10, p = pix & 1023;
  int i = p >> 5, j = p & 31;
  float acc = Bd[ch];
#pragma unroll
  for (int di = 0; di < 3; ++di) {
    int ii = i + di - 1;
    if (ii < 0 || ii > 31) continue;
#pragma unroll
    for (int dj = 0; dj < 3; ++dj) {
      int jj = j + dj - 1;
      if (jj < 0 || jj > 31) continue;
      acc += Wd[ch * 9 + di * 3 + dj] * Vf[((b << 10) + ii * 32 + jj) * 256 + ch];
    }
  }
  O[idx] += acc;
}

// assemble [cls; O] as f16 rows for the projection GEMM
__global__ void k_cast_y(const float* __restrict__ x, const float* __restrict__ O,
                         _Float16* __restrict__ Yh) {
  int idx = blockIdx.x * 256 + threadIdx.x;   // 32800*256
  int cc = idx & 255, row = idx >> 8;
  int b = row / 1025, t = row % 1025;
  float v = (t == 0) ? x[(b * 1025) * 256 + cc]
                     : O[((b << 10) + t - 1) * 256 + cc];
  Yh[idx] = (_Float16)v;
}

// out[32800,256] = Y @ proj_w^T + proj_b
__global__ void k_gemm_proj(const _Float16* __restrict__ Yh, const _Float16* __restrict__ Wp,
                            const float* __restrict__ Bp, float* __restrict__ Out) {
  const int wid = blockIdx.x * 8 + (threadIdx.x >> 5);   // 8200 waves
  const int mt = wid >> 2, nc = wid & 3;
  const int m0 = mt * 16, n0 = nc * 64;
  v8f acc[4] = {};
#pragma unroll
  for (int k0 = 0; k0 < 256; k0 += 32) {
    v16h a = load_frag(Yh + m0 * 256 + k0, 256);
    v16h b[4];
#pragma unroll
    for (int t = 0; t < 4; ++t)
      b[t] = load_frag(Wp + (n0 + t * 16) * 256 + k0, 256);
#pragma unroll
    for (int t = 0; t < 4; ++t)
      acc[t] = wmma16(a, b[t], acc[t]);
  }
  const int l = threadIdx.x & 31, nn = l & 15, mb = (l >> 4) << 3;
#pragma unroll
  for (int t = 0; t < 4; ++t) {
    int ng = n0 + t * 16 + nn;
#pragma unroll
    for (int r = 0; r < 8; ++r)
      Out[(m0 + mb + r) * 256 + ng] = acc[t][r] + Bp[ng];
  }
}

// ---------------------------------------------------------------------------
extern "C" void kernel_launch(void* const* d_in, const int* in_sizes, int n_in,
                              void* d_out, int out_size, void* d_ws, size_t ws_size,
                              hipStream_t stream) {
  (void)in_sizes; (void)n_in; (void)out_size; (void)ws_size;
  const float* x     = (const float*)d_in[0];
  const float* qkvw  = (const float*)d_in[1];
  const float* projw = (const float*)d_in[2];
  const float* projb = (const float*)d_in[3];
  const float* dwcw  = (const float*)d_in[4];
  const float* dwcb  = (const float*)d_in[5];
  const float* anb   = (const float*)d_in[6];
  const float* ahb   = (const float*)d_in[7];
  const float* awb   = (const float*)d_in[8];
  const float* nab   = (const float*)d_in[9];
  const float* hab   = (const float*)d_in[10];
  const float* wab   = (const float*)d_in[11];
  float* out = (float*)d_out;

  char* ws = (char*)d_ws;
  size_t o = 0;
  auto alloc = [&](size_t bytes) { size_t r = o; o += (bytes + 255) & ~(size_t)255; return r; };
  _Float16* Xh  = (_Float16*)(ws + alloc(32768ull * 256 * 2));
  _Float16* Wq  = (_Float16*)(ws + alloc(768ull * 256 * 2));
  _Float16* Wp  = (_Float16*)(ws + alloc(256ull * 256 * 2));
  _Float16* Qh  = (_Float16*)(ws + alloc(32ull * 8 * 1024 * 32 * 2));
  _Float16* Kh  = (_Float16*)(ws + alloc(32ull * 8 * 1024 * 32 * 2));
  _Float16* Vt  = (_Float16*)(ws + alloc(32ull * 8 * 32 * 1024 * 2));
  float*    Qf  = (float*)(ws + alloc(32768ull * 256 * 4));
  float*    Vf  = (float*)(ws + alloc(32768ull * 256 * 4));
  _Float16* Ah  = (_Float16*)(ws + alloc(32ull * 8 * 64 * 32 * 2));
  float*    BA  = (float*)(ws + alloc(8ull * 64 * 1024 * 4));
  float*    BQ  = (float*)(ws + alloc(8ull * 1024 * 64 * 4));
  float*    S   = (float*)(ws + alloc(32ull * 8 * 64 * 1024 * 4));   // reused S1/S2
  _Float16* P   = (_Float16*)(ws + alloc(32ull * 8 * 64 * 1024 * 2)); // reused P1/P2
  _Float16* AVt = (_Float16*)(ws + alloc(32ull * 8 * 32 * 64 * 2));
  _Float16* Yh  = (_Float16*)(ws + alloc(32800ull * 256 * 2));
  float*    O   = Qf;   // reuse q-f32 buffer (only needed by k_agents) as attn-out

  dim3 B(256);
  k_prep_w  <<<768,   B, 0, stream>>>(qkvw, projw, Wq, Wp);
  k_cast_x  <<<32768, B, 0, stream>>>(x, Xh);
  k_gemm_qkv<<<3072,  B, 0, stream>>>(Xh, Wq, Qf, Vf, Qh, Kh, Vt);
  k_agents  <<<2048,  B, 0, stream>>>(Qf, Ah);
  k_biasA   <<<2048,  B, 0, stream>>>(anb, ahb, awb, BA);
  k_biasQ   <<<2048,  B, 0, stream>>>(nab, hab, wab, BQ);
  k_gemm_s1 <<<2048,  B, 0, stream>>>(Ah, Kh, BA, S);
  k_softmax1<<<16384, B, 0, stream>>>(S, P);
  k_gemm_av <<<128,   B, 0, stream>>>(P, Vt, AVt);
  k_gemm_s2 <<<2048,  B, 0, stream>>>(Qh, Ah, BQ, S);
  k_softmax2<<<32768, B, 0, stream>>>(S, P);
  k_gemm_out<<<2048,  B, 0, stream>>>(P, AVt, O);
  k_dwc     <<<32768, B, 0, stream>>>(Vf, dwcw, dwcb, O);
  k_cast_y  <<<32800, B, 0, stream>>>(x, O, Yh);
  k_gemm_proj<<<1025, B, 0, stream>>>(Yh, Wp, projb, out);
}
